// Loss_42056319762459
// MI455X (gfx1250) — compile-verified, matrix operand order checked by reading, not run on hardware
//
#include <hip/hip_runtime.h>
#include <hip/hip_bf16.h>

// CDNA5 / gfx1250, wave32. Loss = CRF-NLL + 2 cross-entropies.
// CRF forward: two-level log-semiring scan; per-step log-matmul done in exp
// space with per-column rescaling via V_WMMA_F32_16X16X32_F16.

typedef __attribute__((ext_vector_type(16))) _Float16 v16h;
typedef __attribute__((ext_vector_type(8)))  float    v8f;

#define KTAG 16

// raw transcendental ops (no denorm/range fixup branches)
__device__ __forceinline__ float fexp(float x) {
  return __builtin_amdgcn_exp2f(x * 1.4426950408889634f);   // v_exp_f32
}
__device__ __forceinline__ float flog(float x) {
  return __builtin_amdgcn_logf(x) * 0.6931471805599453f;    // v_log_f32
}
__device__ __forceinline__ float frcp(float x) {
  return __builtin_amdgcn_rcpf(x);                          // v_rcp_f32
}
// swap between lanes L and L+16: group-of-32 ds_swizzle, xor=0x10, and=0x1f
__device__ __forceinline__ float swz16(float x) {
  return __int_as_float(__builtin_amdgcn_ds_swizzle(__float_as_int(x), 0x401F));
}

// ---------------------------------------------------------------------------
// Level 1: one wave reduces chunk c of batch b to a 16x16 log-transfer matrix
// logM[k_in][k_out] over steps [t0, t1).  State in exp space, normalized per
// column (k_in), log-scale l per column:
//   D[k_out,k_in] = sum_k expT[k,k_out]*state[k,k_in]          (WMMA)
//   u = D * exp(em[t,k_out]);  cm = colmax(u);  l += log cm;  state = u/cm
// The f32 C/D layout (lane=N, VGPR v = rows v/v+8) maps element-for-element
// onto the f16 B layout, so the state feeds the next WMMA with only cvt ops.
// Mask is a prefix mask: valid steps = first `cnt` of the chunk (ballot'd
// once), so the hot loop is branch-free and has no scalar loads.
// ---------------------------------------------------------------------------
__global__ __launch_bounds__(32) void crf_chunk_scan(
    const float* __restrict__ em, const int* __restrict__ mask,
    const float* __restrict__ T, float* __restrict__ logM,
    int S, int C, int Lc) {
  const int c    = blockIdx.x;
  const int b    = blockIdx.y;
  const int lane = threadIdx.x;          // 0..31
  const int col  = lane & 15;            // N index (k_in)
  const int koff = (lane & 16) ? 8 : 0;  // row-half handled by this lane

  // A = exp(T)^T in f16, 16x32 with K=16..31 zero-padded.
  v16h Ar;
  for (int e = 0; e < 16; ++e) Ar[e] = (_Float16)0.0f;
  for (int e = 0; e < 8; ++e)
    Ar[e] = (_Float16)fexp(T[(koff + e) * KTAG + col]);

  const float* emb = em + (size_t)b * S * KTAG;
  const int*   mb  = mask + (size_t)b * S;

  // identity state
  float a[8];
  float l = 0.0f;
  v16h Br;
  for (int e = 0; e < 16; ++e) Br[e] = (_Float16)0.0f;
  for (int e = 0; e < 8; ++e) {
    a[e]  = (koff + e == col) ? 1.0f : 0.0f;
    Br[e] = (_Float16)a[e];
  }

  const int t0 = 1 + c * Lc;
  int t1 = t0 + Lc; if (t1 > S) t1 = S;

  // count valid (masked-in) steps in this chunk; mask is a prefix mask so
  // the valid steps are exactly the first `cnt` steps of the chunk.
  int cnt = 0;
  for (int tt = t0; tt < t1; tt += 32) {
    int tc = tt + lane;
    int mv = (tc < t1) ? mb[tc] : 0;
    unsigned bal = __builtin_amdgcn_ballot_w32(mv != 0);
    cnt += __builtin_popcount(bal);
  }
  const int tend = t0 + cnt;             // wave-uniform

  // software-pipelined emission loads (prefetch t+1, clamped in-bounds)
  float4 e0 = {0, 0, 0, 0}, e1 = {0, 0, 0, 0};
  if (t0 < tend) {
    const float* ep = emb + (size_t)t0 * KTAG + koff;
    e0 = *(const float4*)ep;
    e1 = *(const float4*)(ep + 4);
  }
  for (int t = t0; t < tend; ++t) {
    float4 c0 = e0, c1 = e1;
    int tn = t + 1; if (tn > S - 1) tn = S - 1;
    const float* epn = emb + (size_t)tn * KTAG + koff;
    e0 = *(const float4*)epn;
    e1 = *(const float4*)(epn + 4);

    v8f cz = {};
    v8f p = __builtin_amdgcn_wmma_f32_16x16x32_f16(
        false, Ar, false, Br, (short)0, cz, false, false);

    float u[8];
    u[0] = p[0] * fexp(c0.x);
    u[1] = p[1] * fexp(c0.y);
    u[2] = p[2] * fexp(c0.z);
    u[3] = p[3] * fexp(c0.w);
    u[4] = p[4] * fexp(c1.x);
    u[5] = p[5] * fexp(c1.y);
    u[6] = p[6] * fexp(c1.z);
    u[7] = p[7] * fexp(c1.w);

    float cm = fmaxf(fmaxf(fmaxf(u[0], u[1]), fmaxf(u[2], u[3])),
                     fmaxf(fmaxf(u[4], u[5]), fmaxf(u[6], u[7])));
    cm = fmaxf(cm, swz16(cm));           // full column max across lane pair
    float inv = frcp(cm);

    l += flog(cm);
    for (int e = 0; e < 8; ++e) {
      a[e]  = u[e] * inv;
      Br[e] = (_Float16)a[e];
    }
  }

  // logM[b][c][k_in=col][k_out=koff+e] = l + log a[e]   (two b128 stores)
  float* op = logM + (((size_t)b * C + c) * KTAG + col) * KTAG + koff;
  float4 o0, o1;
  o0.x = l + flog(a[0]); o0.y = l + flog(a[1]);
  o0.z = l + flog(a[2]); o0.w = l + flog(a[3]);
  o1.x = l + flog(a[4]); o1.y = l + flog(a[5]);
  o1.z = l + flog(a[6]); o1.w = l + flog(a[7]);
  *(float4*)op       = o0;
  *(float4*)(op + 4) = o1;
}

// ---------------------------------------------------------------------------
// Level 2: per batch, alpha0 = start + em[:,0], fold C chunk matrices
// (log-semiring vec-mat), finish with logZ = LSE(alpha + end).
// ---------------------------------------------------------------------------
__global__ __launch_bounds__(32) void crf_combine(
    const float* __restrict__ em, const float* __restrict__ logM,
    const float* __restrict__ startT, const float* __restrict__ endT,
    float* __restrict__ logZ, int S, int C) {
  const int b = blockIdx.x;
  const int lane = threadIdx.x;
  __shared__ float al[KTAG];
  if (lane < KTAG) al[lane] = startT[lane] + em[(size_t)b * S * KTAG + lane];
  __syncthreads();
  for (int c = 0; c < C; ++c) {
    const float* Mc = logM + ((size_t)b * C + c) * (KTAG * KTAG);
    float nv = 0.0f;
    if (lane < KTAG) {
      float x[KTAG];
      float mx = -3.0e38f;
      for (int k = 0; k < KTAG; ++k) {
        x[k] = al[k] + Mc[k * KTAG + lane];
        mx = fmaxf(mx, x[k]);
      }
      float se = 0.0f;
      for (int k = 0; k < KTAG; ++k) se += fexp(x[k] - mx);
      nv = mx + flog(se);
    }
    __syncthreads();
    if (lane < KTAG) al[lane] = nv;
    __syncthreads();
  }
  if (lane == 0) {
    float x[KTAG];
    float mx = -3.0e38f;
    for (int k = 0; k < KTAG; ++k) { x[k] = al[k] + endT[k]; mx = fmaxf(mx, x[k]); }
    float se = 0.0f;
    for (int k = 0; k < KTAG; ++k) se += fexp(x[k] - mx);
    logZ[b] = mx + flog(se);
  }
}

// ---------------------------------------------------------------------------
// CRF numerator (gold-path score), one block per batch
// ---------------------------------------------------------------------------
__global__ __launch_bounds__(256) void crf_num_kernel(
    const float* __restrict__ em, const int* __restrict__ mask,
    const int* __restrict__ tags, const float* __restrict__ T,
    const float* __restrict__ startT, const float* __restrict__ endT,
    float* __restrict__ num, int S) {
  const int b = blockIdx.x;
  const int tid = threadIdx.x;
  const float* emr = em + (size_t)b * S * KTAG;
  const int* mr = mask + (size_t)b * S;
  const int* tr = tags + (size_t)b * S;
  float acc = 0.0f, cnt = 0.0f;
  for (int t = tid; t < S; t += 256) {
    if (mr[t] != 0) {
      cnt += 1.0f;
      if (t >= 1) {
        int tp = tr[t - 1], tc = tr[t];
        acc += T[tp * KTAG + tc] + emr[(size_t)t * KTAG + tc];
      }
    }
  }
  __shared__ float sa[256];
  __shared__ float sc[256];
  sa[tid] = acc; sc[tid] = cnt;
  __syncthreads();
  for (int s = 128; s > 0; s >>= 1) {
    if (tid < s) { sa[tid] += sa[tid + s]; sc[tid] += sc[tid + s]; }
    __syncthreads();
  }
  if (tid == 0) {
    int len = (int)(sc[0] + 0.5f);
    int tg0 = tr[0];
    num[b] = sa[0] + startT[tg0] + emr[tg0] + endT[tr[len - 1]];
  }
}

// ---------------------------------------------------------------------------
// Cross-entropies + reductions
// ---------------------------------------------------------------------------
__global__ void zero_sums(float* sums) {
  if (threadIdx.x < 3) sums[threadIdx.x] = 0.0f;
}

__global__ __launch_bounds__(256) void ce_entity_kernel(
    const float* __restrict__ logits, const int* __restrict__ labels,
    float* __restrict__ sums, int rows, int Kc) {
  int r = blockIdx.x * 256 + threadIdx.x;
  float nll = 0.0f, val = 0.0f;
  if (r < rows) {
    const float* row = logits + (size_t)r * Kc;
    float mx = row[0];
    for (int k = 1; k < Kc; ++k) mx = fmaxf(mx, row[k]);
    float se = 0.0f;
    for (int k = 0; k < Kc; ++k) se += fexp(row[k] - mx);
    int lb = labels[r];
    float n = (mx + flog(se)) - row[lb];
    val = (lb != 0) ? 1.0f : 0.0f;
    nll = n * val;
  }
  __shared__ float sa[256];
  __shared__ float sv[256];
  sa[threadIdx.x] = nll; sv[threadIdx.x] = val;
  __syncthreads();
  for (int s = 128; s > 0; s >>= 1) {
    if (threadIdx.x < s) {
      sa[threadIdx.x] += sa[threadIdx.x + s];
      sv[threadIdx.x] += sv[threadIdx.x + s];
    }
    __syncthreads();
  }
  if (threadIdx.x == 0) {
    atomicAdd(&sums[0], sa[0]);
    atomicAdd(&sums[1], sv[0]);
  }
}

__global__ __launch_bounds__(256) void ce_intent_kernel(
    const float* __restrict__ logits, const int* __restrict__ labels,
    float* __restrict__ sums, int B, int NI) {
  int tid = threadIdx.x;
  float nll = 0.0f;
  for (int r = tid; r < B; r += 256) {
    const float* row = logits + (size_t)r * NI;
    float mx = row[0];
    for (int k = 1; k < NI; ++k) mx = fmaxf(mx, row[k]);
    float se = 0.0f;
    for (int k = 0; k < NI; ++k) se += fexp(row[k] - mx);
    nll += (mx + flog(se)) - row[labels[r]];
  }
  __shared__ float sa[256];
  sa[tid] = nll;
  __syncthreads();
  for (int s = 128; s > 0; s >>= 1) {
    if (tid < s) sa[tid] += sa[tid + s];
    __syncthreads();
  }
  if (tid == 0) sums[2] = sa[0];
}

__global__ __launch_bounds__(256) void finalize_kernel(
    const float* __restrict__ logZ, const float* __restrict__ num,
    const float* __restrict__ sums, float* __restrict__ out, int B) {
  int tid = threadIdx.x;
  float v = 0.0f;
  for (int b = tid; b < B; b += 256) v += logZ[b] - num[b];
  __shared__ float sa[256];
  sa[tid] = v;
  __syncthreads();
  for (int s = 128; s > 0; s >>= 1) {
    if (tid < s) sa[tid] += sa[tid + s];
    __syncthreads();
  }
  if (tid == 0) {
    float e1 = sa[0] / (float)B;
    float e2 = sums[0] / fmaxf(sums[1], 1.0f);
    float li = sums[2] / (float)B;
    out[0] = (e1 + e2 + li) / 3.0f;
    out[1] = e1;
    out[2] = e2;
    out[3] = li;
  }
}

// ---------------------------------------------------------------------------
extern "C" void kernel_launch(void* const* d_in, const int* in_sizes, int n_in,
                              void* d_out, int out_size, void* d_ws, size_t ws_size,
                              hipStream_t stream) {
  const float* em        = (const float*)d_in[0];   // (B,S,K) f32
  const int*   mask      = (const int*)d_in[1];     // (B,S)   bool->int
  const float* ent_logit = (const float*)d_in[2];   // (B,E,KE)
  const float* int_logit = (const float*)d_in[3];   // (B,NI)
  const int*   seq_lab   = (const int*)d_in[4];     // (B,S)
  const int*   ent_lab   = (const int*)d_in[5];     // (B,E)
  const int*   int_lab   = (const int*)d_in[6];     // (B,)
  const float* T         = (const float*)d_in[7];   // (K,K)
  const float* startT    = (const float*)d_in[8];   // (K,)
  const float* endT      = (const float*)d_in[9];   // (K,)

  const int B     = in_sizes[6];
  const int S     = in_sizes[1] / B;
  const int NI    = in_sizes[3] / B;
  const int rowsE = in_sizes[5];
  const int KE    = in_sizes[2] / rowsE;

  float* ws   = (float*)d_ws;
  float* logZ = ws;               // [B]
  float* num  = ws + B;           // [B]
  float* sums = ws + 2 * B;       // [3]
  float* logM = ws + 1024;        // [B*C*256]

  // chunk count from available workspace (deterministic for fixed ws_size)
  size_t avail = (ws_size / 4 > 1024) ? (ws_size / 4 - 1024) : 0;
  int C = (int)(avail / ((size_t)B * KTAG * KTAG));
  if (C < 1)  C = 1;
  if (C > 64) C = 64;
  const int Lc = (S - 1 + C - 1) / C;

  zero_sums<<<1, 32, 0, stream>>>(sums);

  dim3 g1(C, B);
  crf_chunk_scan<<<g1, 32, 0, stream>>>(em, mask, T, logM, S, C, Lc);
  crf_combine<<<B, 32, 0, stream>>>(em, logM, startT, endT, logZ, S, C);
  crf_num_kernel<<<B, 256, 0, stream>>>(em, mask, seq_lab, T, startT, endT, num, S);

  ce_entity_kernel<<<(rowsE + 255) / 256, 256, 0, stream>>>(ent_logit, ent_lab, sums, rowsE, KE);
  ce_intent_kernel<<<1, 256, 0, stream>>>(int_logit, int_lab, sums, B, NI);

  finalize_kernel<<<1, 256, 0, stream>>>(logZ, num, sums, (float*)d_out, B);
  (void)n_in; (void)out_size;
}
